// RotationInvariantFeat_35493609734278
// MI455X (gfx1250) — matrix operand order repaired and checked by reading, not compiled, not run on hardware
//
#include <hip/hip_runtime.h>
#include <hip/hip_bf16.h>

#define NPTS 8192
#define KNN  64
#define EPSF 1e-7f
#define BNEPS 1e-5f

typedef __bf16 bf16_t;
typedef __attribute__((ext_vector_type(16))) __bf16 v16bf;
typedef __attribute__((ext_vector_type(8)))  float   v8f;

// ---------------------------------------------------------------------------
// WMMA helpers (CDNA5 wave32, D = A(16x32 bf16) x B(32x16 bf16) + C(16x16 f32))
// ---------------------------------------------------------------------------
static __device__ inline v8f wmma_bf16(v16bf a, v16bf b, v8f c) {
  return __builtin_amdgcn_wmma_f32_16x16x32_bf16(false, a, false, b, (short)0, c,
                                                 false, false);
}

// A-fragment from a row-major bf16 row (stride >= k0+32).
// ISA 16-bit A 16x32 layout: lane<16 holds K = k0+{0..7, 16..23} of row M=lane;
// lane>=16 holds K = k0+{8..15, 24..31} of row M=lane-16. Two contiguous 16B chunks.
static __device__ inline v16bf load_a_frag(const bf16_t* row, int k0, int alo) {
  v16bf f;
  const bf16_t* p0 = row + k0 + alo;
  const bf16_t* p1 = row + k0 + 16 + alo;
#pragma unroll
  for (int i = 0; i < 8; ++i) { f[i] = p0[i]; f[i + 8] = p1[i]; }
  return f;
}

// B-fragment from pre-packed weights: 16 contiguous bf16 per lane.
static __device__ inline v16bf load_b_frag(const bf16_t* p) {
  v16bf f;
#pragma unroll
  for (int i = 0; i < 16; ++i) f[i] = p[i];
  return f;
}

// ---------------------------------------------------------------------------
// 1) KNN: one block per query point; 32KB LDS distance array; 64 argmin rounds
// ---------------------------------------------------------------------------
__global__ void knn_topk_kernel(const float* __restrict__ pts, int* __restrict__ idx) {
  __shared__ float dist[NPTS];
  __shared__ float vred[256];
  __shared__ int   ired[256];
  const int n = blockIdx.x;
  const int tid = threadIdx.x;
  const float qx = pts[n * 3 + 0], qy = pts[n * 3 + 1], qz = pts[n * 3 + 2];
  for (int m = tid; m < NPTS; m += 256) {
    float dx = pts[m * 3 + 0] - qx, dy = pts[m * 3 + 1] - qy, dz = pts[m * 3 + 2] - qz;
    dist[m] = dx * dx + dy * dy + dz * dz;
  }
  __syncthreads();
  for (int kk = 0; kk < KNN; ++kk) {
    float best = 3.4e38f; int bi = NPTS;
    for (int m = tid; m < NPTS; m += 256) {
      float d = dist[m];
      if (d < best || (d == best && m < bi)) { best = d; bi = m; }
    }
    vred[tid] = best; ired[tid] = bi;
    __syncthreads();
    for (int s = 128; s > 0; s >>= 1) {
      if (tid < s) {
        float vo = vred[tid + s]; int io = ired[tid + s];
        if (vo < vred[tid] || (vo == vred[tid] && io < ired[tid])) {
          vred[tid] = vo; ired[tid] = io;
        }
      }
      __syncthreads();
    }
    if (tid == 0) { idx[n * KNN + kk] = ired[0]; dist[ired[0]] = 3.4e38f; }
    __syncthreads();
  }
}

// ---------------------------------------------------------------------------
// 2) Local rotation-invariant features -> featup bf16 [N*K, 32] (cols 4..31 = 0)
// ---------------------------------------------------------------------------
__global__ void proj_knn_kernel(const float* __restrict__ pts, const int* __restrict__ idx,
                                bf16_t* __restrict__ featup) {
  __shared__ float pcx[KNN], pcy[KNN], pcz[KNN], vns[KNN];
  __shared__ float ax[9];
  const int n = blockIdx.x, k = threadIdx.x;
  const int i0 = idx[n * KNN + 0];
  const float cx = pts[i0 * 3], cy = pts[i0 * 3 + 1], cz = pts[i0 * 3 + 2];
  const int ik = idx[n * KNN + k];
  float px = pts[ik * 3] - cx, py = pts[ik * 3 + 1] - cy, pz = pts[ik * 3 + 2] - cz;
  float vn = sqrtf(px * px + py * py + pz * pz);
  pcx[k] = px; pcy[k] = py; pcz[k] = pz; vns[k] = vn;
  __syncthreads();
  if (k == 0) {
    int id1 = 0; float best = vns[0];
    for (int j = 1; j < KNN; ++j) if (vns[j] > best) { best = vns[j]; id1 = j; }
    float a1x = pcx[id1], a1y = pcy[id1], a1z = pcz[id1];
    float n1 = sqrtf(a1x * a1x + a1y * a1y + a1z * a1z) + EPSF;
    a1x /= n1; a1y /= n1; a1z /= n1;
    float mx = 0.f, my = 0.f, mz = 0.f;
    for (int j = 0; j < KNN; ++j) { mx += pcx[j]; my += pcy[j]; mz += pcz[j]; }
    mx /= (float)KNN; my /= (float)KNN; mz /= (float)KNN;
    float n2 = sqrtf(mx * mx + my * my + mz * mz) + EPSF;
    float a2x = mx / n2, a2y = my / n2, a2z = mz / n2;
    float a3x = a1x + 1.5f * a2x, a3y = a1y + 1.5f * a2y, a3z = a1z + 1.5f * a2z;
    float n3 = sqrtf(a3x * a3x + a3y * a3y + a3z * a3z) + EPSF;
    a3x /= n3; a3y /= n3; a3z /= n3;
    ax[0] = a1x; ax[1] = a1y; ax[2] = a1z;
    ax[3] = a2x; ax[4] = a2y; ax[5] = a2z;
    ax[6] = a3x; ax[7] = a3y; ax[8] = a3z;
  }
  __syncthreads();
  float den = vn + EPSF;
  float f1 = (px * ax[0] + py * ax[1] + pz * ax[2]) / den;
  float f2 = (px * ax[3] + py * ax[4] + pz * ax[5]) / den;
  float f3 = (px * ax[6] + py * ax[7] + pz * ax[8]) / den;
  bf16_t* row = featup + (size_t)(n * KNN + k) * 32;
  row[0] = (bf16_t)f1; row[1] = (bf16_t)f2; row[2] = (bf16_t)f3; row[3] = (bf16_t)vn;
#pragma unroll
  for (int j = 4; j < 32; ++j) row[j] = (bf16_t)0.f;
}

// ---------------------------------------------------------------------------
// 3) Global projection axes (argmax/argmin of |p|) -> avecs[9]
// ---------------------------------------------------------------------------
__global__ void proj_points_setup_kernel(const float* __restrict__ pts,
                                         float* __restrict__ avecs) {
  __shared__ float vmx[256], vmn[256];
  __shared__ int   imx[256], imn[256];
  const int tid = threadIdx.x;
  float bmax = -1.f, bmin = 3.4e38f; int bi = 0, bj = 0;
  for (int n = tid; n < NPTS; n += 256) {
    float x = pts[n * 3], y = pts[n * 3 + 1], z = pts[n * 3 + 2];
    float v = sqrtf(x * x + y * y + z * z);
    if (v > bmax) { bmax = v; bi = n; }
    if (v < bmin) { bmin = v; bj = n; }
  }
  vmx[tid] = bmax; imx[tid] = bi; vmn[tid] = bmin; imn[tid] = bj;
  __syncthreads();
  for (int s = 128; s > 0; s >>= 1) {
    if (tid < s) {
      if (vmx[tid + s] > vmx[tid] || (vmx[tid + s] == vmx[tid] && imx[tid + s] < imx[tid])) {
        vmx[tid] = vmx[tid + s]; imx[tid] = imx[tid + s];
      }
      if (vmn[tid + s] < vmn[tid] || (vmn[tid + s] == vmn[tid] && imn[tid + s] < imn[tid])) {
        vmn[tid] = vmn[tid + s]; imn[tid] = imn[tid + s];
      }
    }
    __syncthreads();
  }
  if (tid == 0) {
    int i1 = imx[0], i2 = imn[0];
    float a1x = pts[i1 * 3], a1y = pts[i1 * 3 + 1], a1z = pts[i1 * 3 + 2];
    float n1 = sqrtf(a1x * a1x + a1y * a1y + a1z * a1z) + EPSF;
    a1x /= n1; a1y /= n1; a1z /= n1;
    float a2x = pts[i2 * 3], a2y = pts[i2 * 3 + 1], a2z = pts[i2 * 3 + 2];
    float n2 = sqrtf(a2x * a2x + a2y * a2y + a2z * a2z) + EPSF;
    a2x /= n2; a2y /= n2; a2z /= n2;
    float a3x = a1x + 1.5f * a2x, a3y = a1y + 1.5f * a2y, a3z = a1z + 1.5f * a2z;
    float n3 = sqrtf(a3x * a3x + a3y * a3y + a3z * a3z) + EPSF;
    a3x /= n3; a3y /= n3; a3z /= n3;
    avecs[0] = a1x; avecs[1] = a1y; avecs[2] = a1z;
    avecs[3] = a2x; avecs[4] = a2y; avecs[5] = a2z;
    avecs[6] = a3x; avecs[7] = a3y; avecs[8] = a3z;
  }
}

// feat_down bf16 [N, 32] (cols 4..31 = 0)
__global__ void featdown_kernel(const float* __restrict__ pts, const float* __restrict__ a,
                                bf16_t* __restrict__ fd) {
  int n = blockIdx.x * blockDim.x + threadIdx.x;
  if (n >= NPTS) return;
  float px = pts[n * 3], py = pts[n * 3 + 1], pz = pts[n * 3 + 2];
  float vn = sqrtf(px * px + py * py + pz * pz);
  float den = vn + EPSF;
  float f1 = (px * a[0] + py * a[1] + pz * a[2]) / den;
  float f2 = (px * a[3] + py * a[4] + pz * a[5]) / den;
  float f3 = (px * a[6] + py * a[7] + pz * a[8]) / den;
  bf16_t* row = fd + (size_t)n * 32;
  row[0] = (bf16_t)f1; row[1] = (bf16_t)f2; row[2] = (bf16_t)f3; row[3] = (bf16_t)vn;
#pragma unroll
  for (int j = 4; j < 32; ++j) row[j] = (bf16_t)0.f;
}

// ---------------------------------------------------------------------------
// 4) Pack f32 weights into WMMA B-fragment-contiguous bf16 layout.
//    B[k][n] = src[k*sk + n*sn]; zero-pad k >= Kreal.
//    dst[((nb*KB+kb)*32 + lane)*16 + e] with k = kb*32 + (lane<16?0:8) + (e<8?e:e+8),
//    n = nb*16 + (lane&15).
// ---------------------------------------------------------------------------
__global__ void pack_b_kernel(const float* __restrict__ src, bf16_t* __restrict__ dst,
                              int Kreal, int Kpad, int Nc, int sk, int sn) {
  int e = blockIdx.x * blockDim.x + threadIdx.x;
  int total = Kpad * Nc;
  if (e >= total) return;
  int tile = e >> 9, within = e & 511;
  int lane = within >> 4, el = within & 15;
  int KB = Kpad / 32;
  int kb = tile % KB, nb = tile / KB;
  int k = kb * 32 + ((lane < 16) ? 0 : 8) + (el < 8 ? el : el + 8);
  int nn = nb * 16 + (lane & 15);
  float v = (k < Kreal) ? src[k * sk + nn * sn] : 0.f;
  dst[e] = (bf16_t)v;
}

// ---------------------------------------------------------------------------
// 5) Fused 5-layer MLP (4->64 x5, ReLU) + max over K=64, all in WMMA.
//    Block = 256 threads = 8 waves = 2 points (4 waves of 16 rows each).
// ---------------------------------------------------------------------------
__global__ void mlp_fused_kernel(const bf16_t* __restrict__ featup,
                                 const bf16_t* __restrict__ w0p,
                                 const bf16_t* __restrict__ w1p,
                                 const bf16_t* __restrict__ w2p,
                                 const bf16_t* __restrict__ w3p,
                                 const bf16_t* __restrict__ w4p,
                                 bf16_t* __restrict__ concatBuf) {
  __shared__ bf16_t stage[8 * 16 * 64];  // per-wave 16x64 activation staging
  const int wave = threadIdx.x >> 5;
  const int lane = threadIdx.x & 31;
  const int point = blockIdx.x * 2 + (wave >> 2);
  const int sub = wave & 3;
  const size_t rowBase = (size_t)point * KNN + sub * 16;
  bf16_t* my = stage + wave * 1024;
  const int hl = lane >> 4;
  const int mrow = lane & 15;
  const int alo = hl ? 8 : 0;

  v8f acc[4];
  // ---- layer 0: featup [NK,32] x w0 [32pad,64] ----
  {
    const bf16_t* ar = featup + (rowBase + mrow) * 32;
    v16bf a = load_a_frag(ar, 0, alo);
#pragma unroll
    for (int nb = 0; nb < 4; ++nb) {
      v16bf b = load_b_frag(w0p + (size_t)nb * 512 + lane * 16);
      v8f c = {};
      acc[nb] = wmma_bf16(a, b, c);
    }
#pragma unroll
    for (int nb = 0; nb < 4; ++nb)
#pragma unroll
      for (int r = 0; r < 8; ++r)
        my[(r + 8 * hl) * 64 + nb * 16 + mrow] = (bf16_t)fmaxf(acc[nb][r], 0.f);
  }
  __syncthreads();
  // ---- layers 1..4: [16,64] x [64,64] ----
  const bf16_t* Ws[4] = {w1p, w2p, w3p, w4p};
#pragma unroll
  for (int l = 0; l < 4; ++l) {
    const bf16_t* ar = my + mrow * 64;
    v16bf a0 = load_a_frag(ar, 0, alo);
    v16bf a1 = load_a_frag(ar, 32, alo);
    __syncthreads();  // all reads of `stage` done before overwrite
#pragma unroll
    for (int nb = 0; nb < 4; ++nb) {
      v16bf b0 = load_b_frag(Ws[l] + (size_t)(nb * 2 + 0) * 512 + lane * 16);
      v16bf b1 = load_b_frag(Ws[l] + (size_t)(nb * 2 + 1) * 512 + lane * 16);
      v8f c = {};
      c = wmma_bf16(a0, b0, c);
      c = wmma_bf16(a1, b1, c);
      acc[nb] = c;
    }
#pragma unroll
    for (int nb = 0; nb < 4; ++nb)
#pragma unroll
      for (int r = 0; r < 8; ++r)
        my[(r + 8 * hl) * 64 + nb * 16 + mrow] = (bf16_t)fmaxf(acc[nb][r], 0.f);
    __syncthreads();
  }
  // ---- max over K=64 rows per point -> concatBuf cols 0..63 (stride 192) ----
  const int tid = threadIdx.x;
  if (tid < 128) {
    int p = tid >> 6, c = tid & 63;
    float m = 0.f;  // post-ReLU values are >= 0
    for (int r = 0; r < KNN; ++r)
      m = fmaxf(m, (float)stage[(p * 4 + (r >> 4)) * 1024 + (r & 15) * 64 + c]);
    concatBuf[(size_t)(blockIdx.x * 2 + p) * 192 + c] = (bf16_t)m;
  }
}

// ---------------------------------------------------------------------------
// 6) Generic WMMA GEMM, 16x64 tile per wave (A-fragment reused by 4 WMMAs per
//    K-step): C_f32[M,Nc] = A_bf16[M,lda(>=Kpad)] x Bp + bias.  Nc % 64 == 0.
// ---------------------------------------------------------------------------
__global__ void gemm_wmma_kernel(const bf16_t* __restrict__ A, int lda,
                                 const bf16_t* __restrict__ Bp,
                                 const float* __restrict__ bias,
                                 float* __restrict__ C, int M, int Kpad, int Nc) {
  const int wave = threadIdx.x >> 5;
  const int lane = threadIdx.x & 31;
  const int tilesN = Nc >> 6;          // 64-wide N tiles
  const int tilesM = M >> 4;
  const int tile = blockIdx.x * 8 + wave;
  if (tile >= tilesM * tilesN) return;
  const int tm = tile / tilesN, tn = tile % tilesN;
  const int KB = Kpad >> 5;
  const int hl = lane >> 4;
  const int alo = hl ? 8 : 0;
  v8f acc[4];
#pragma unroll
  for (int j = 0; j < 4; ++j) { v8f z = {}; acc[j] = z; }
  const bf16_t* arow = A + (size_t)(tm * 16 + (lane & 15)) * lda;
  for (int kb = 0; kb < KB; ++kb) {
    v16bf a = load_a_frag(arow, kb * 32, alo);
#pragma unroll
    for (int j = 0; j < 4; ++j) {
      v16bf b = load_b_frag(Bp + (size_t)((tn * 4 + j) * KB + kb) * 512 + lane * 16);
      acc[j] = wmma_bf16(a, b, acc[j]);
    }
  }
  const int rbase = tm * 16 + 8 * hl;
#pragma unroll
  for (int j = 0; j < 4; ++j) {
    const int col = tn * 64 + j * 16 + (lane & 15);
    const float bv = bias ? bias[col] : 0.f;
#pragma unroll
    for (int r = 0; r < 8; ++r)
      C[(size_t)(rbase + r) * Nc + col] = acc[j][r] + bv;
  }
}

// ---------------------------------------------------------------------------
// 7) BatchNorm (training-mode batch stats) helpers
// ---------------------------------------------------------------------------
__global__ void bn_stats_kernel(const float* __restrict__ X, int M, int C,
                                float* __restrict__ mu, float* __restrict__ inv) {
  __shared__ float rs[256], rq[256];
  const int c = blockIdx.x, tid = threadIdx.x;
  float s = 0.f, q = 0.f;
  for (int m = tid; m < M; m += 256) {
    float v = X[(size_t)m * C + c];
    s += v; q += v * v;
  }
  rs[tid] = s; rq[tid] = q;
  __syncthreads();
  for (int st = 128; st > 0; st >>= 1) {
    if (tid < st) { rs[tid] += rs[tid + st]; rq[tid] += rq[tid + st]; }
    __syncthreads();
  }
  if (tid == 0) {
    float mean = rs[0] / (float)M;
    float var = rq[0] / (float)M - mean * mean;
    mu[c] = mean;
    inv[c] = rsqrtf(var + BNEPS);
  }
}

__global__ void bn_relu_bf16_kernel(const float* __restrict__ X,
                                    const float* __restrict__ g, const float* __restrict__ b,
                                    const float* __restrict__ mu, const float* __restrict__ inv,
                                    bf16_t* __restrict__ dst, int M, int C, int ldd, int coff) {
  int i = blockIdx.x * blockDim.x + threadIdx.x;
  if (i >= M * C) return;
  int m = i / C, c = i % C;
  float y = fmaxf(0.f, g[c] * (X[i] - mu[c]) * inv[c] + b[c]);
  dst[(size_t)m * ldd + coff + c] = (bf16_t)y;
}

__global__ void bn_relu_out_kernel(const float* __restrict__ X,
                                   const float* __restrict__ g, const float* __restrict__ b,
                                   const float* __restrict__ mu, const float* __restrict__ inv,
                                   float* __restrict__ out, int M, int C) {
  int i = blockIdx.x * blockDim.x + threadIdx.x;
  if (i >= M * C) return;
  int m = i / C, c = i % C;
  float y = fmaxf(0.f, g[c] * (X[i] - mu[c]) * inv[c] + b[c]);
  out[(size_t)c * M + m] = y;  // [1,1024,N,1] channel-major
}

// ---------------------------------------------------------------------------
// Host launch
// ---------------------------------------------------------------------------
extern "C" void kernel_launch(void* const* d_in, const int* in_sizes, int n_in,
                              void* d_out, int out_size, void* d_ws, size_t ws_size,
                              hipStream_t stream) {
  (void)in_sizes; (void)n_in; (void)out_size; (void)ws_size;
  const float* points = (const float*)d_in[0];
  const float* gw[5] = {(const float*)d_in[1], (const float*)d_in[2], (const float*)d_in[3],
                        (const float*)d_in[4], (const float*)d_in[5]};
  const float* c1w = (const float*)d_in[6];  const float* c1b = (const float*)d_in[7];
  const float* bn1g = (const float*)d_in[8]; const float* bn1b = (const float*)d_in[9];
  const float* c2w = (const float*)d_in[10]; const float* c2b = (const float*)d_in[11];
  const float* bn2g = (const float*)d_in[12];const float* bn2b = (const float*)d_in[13];
  const float* c3w = (const float*)d_in[14]; const float* c3b = (const float*)d_in[15];
  const float* bn3g = (const float*)d_in[16];const float* bn3b = (const float*)d_in[17];
  const float* c4w = (const float*)d_in[18]; const float* c4b = (const float*)d_in[19];
  const float* bn4g = (const float*)d_in[20];const float* bn4b = (const float*)d_in[21];
  float* out = (float*)d_out;

  // workspace carving
  char* ws = (char*)d_ws;
  size_t off = 0;
  auto carve = [&](size_t bytes) -> char* {
    char* p = ws + off;
    off = (off + bytes + 255) & ~(size_t)255;
    return p;
  };
  int*    idx      = (int*)carve((size_t)NPTS * KNN * 4);
  bf16_t* featup   = (bf16_t*)carve((size_t)NPTS * KNN * 32 * 2);
  float*  avecs    = (float*)carve(16 * 4);
  bf16_t* featdown = (bf16_t*)carve((size_t)NPTS * 32 * 2);
  bf16_t* gw0p = (bf16_t*)carve(32 * 64 * 2);
  bf16_t* gw1p = (bf16_t*)carve(64 * 64 * 2);
  bf16_t* gw2p = (bf16_t*)carve(64 * 64 * 2);
  bf16_t* gw3p = (bf16_t*)carve(64 * 64 * 2);
  bf16_t* gw4p = (bf16_t*)carve(64 * 64 * 2);
  bf16_t* c1p  = (bf16_t*)carve(32 * 64 * 2);
  bf16_t* c2p  = (bf16_t*)carve(64 * 128 * 2);
  bf16_t* c3p  = (bf16_t*)carve(192 * 256 * 2);
  bf16_t* c4p  = (bf16_t*)carve(256 * 1024 * 2);
  float*  convbuf = (float*)carve((size_t)NPTS * 1024 * 4);  // reused per conv
  bf16_t* h1      = (bf16_t*)carve((size_t)NPTS * 64 * 2);
  bf16_t* concatB = (bf16_t*)carve((size_t)NPTS * 192 * 2);
  bf16_t* h3      = (bf16_t*)carve((size_t)NPTS * 256 * 2);
  float*  mu      = (float*)carve(1024 * 4);
  float*  inv     = (float*)carve(1024 * 4);

  // ---- upper branch ----
  knn_topk_kernel<<<NPTS, 256, 0, stream>>>(points, idx);
  proj_knn_kernel<<<NPTS, KNN, 0, stream>>>(points, idx, featup);

  // ---- lower branch features ----
  proj_points_setup_kernel<<<1, 256, 0, stream>>>(points, avecs);
  featdown_kernel<<<NPTS / 256, 256, 0, stream>>>(points, avecs, featdown);

  // ---- pack weights: B[k][n]; gw are [K,N] (sk=N,sn=1), conv w are [N,K] (sk=1,sn=K)
  pack_b_kernel<<<(32 * 64 + 255) / 256, 256, 0, stream>>>(gw[0], gw0p, 4, 32, 64, 64, 1);
  pack_b_kernel<<<(64 * 64 + 255) / 256, 256, 0, stream>>>(gw[1], gw1p, 64, 64, 64, 64, 1);
  pack_b_kernel<<<(64 * 64 + 255) / 256, 256, 0, stream>>>(gw[2], gw2p, 64, 64, 64, 64, 1);
  pack_b_kernel<<<(64 * 64 + 255) / 256, 256, 0, stream>>>(gw[3], gw3p, 64, 64, 64, 64, 1);
  pack_b_kernel<<<(64 * 64 + 255) / 256, 256, 0, stream>>>(gw[4], gw4p, 64, 64, 64, 64, 1);
  pack_b_kernel<<<(32 * 64 + 255) / 256, 256, 0, stream>>>(c1w, c1p, 4, 32, 64, 1, 4);
  pack_b_kernel<<<(64 * 128 + 255) / 256, 256, 0, stream>>>(c2w, c2p, 64, 64, 128, 1, 64);
  pack_b_kernel<<<(192 * 256 + 255) / 256, 256, 0, stream>>>(c3w, c3p, 192, 192, 256, 1, 192);
  pack_b_kernel<<<(256 * 1024 + 255) / 256, 256, 0, stream>>>(c4w, c4p, 256, 256, 1024, 1, 256);

  // ---- fused 5-layer WMMA MLP + max over K -> concat cols [0,64) ----
  mlp_fused_kernel<<<NPTS / 2, 256, 0, stream>>>(featup, gw0p, gw1p, gw2p, gw3p, gw4p,
                                                 concatB);

  // ---- conv1: 4(pad32)->64, BN, ReLU ----  tiles = 512*1
  gemm_wmma_kernel<<<(512 * 1 + 7) / 8, 256, 0, stream>>>(featdown, 32, c1p, c1b, convbuf,
                                                          NPTS, 32, 64);
  bn_stats_kernel<<<64, 256, 0, stream>>>(convbuf, NPTS, 64, mu, inv);
  bn_relu_bf16_kernel<<<(NPTS * 64 + 255) / 256, 256, 0, stream>>>(convbuf, bn1g, bn1b, mu,
                                                                   inv, h1, NPTS, 64, 64, 0);
  // ---- conv2: 64->128, BN, ReLU -> concat cols [64,192) ----  tiles = 512*2
  gemm_wmma_kernel<<<(512 * 2 + 7) / 8, 256, 0, stream>>>(h1, 64, c2p, c2b, convbuf, NPTS,
                                                          64, 128);
  bn_stats_kernel<<<128, 256, 0, stream>>>(convbuf, NPTS, 128, mu, inv);
  bn_relu_bf16_kernel<<<(NPTS * 128 + 255) / 256, 256, 0, stream>>>(convbuf, bn2g, bn2b, mu,
                                                                    inv, concatB, NPTS, 128,
                                                                    192, 64);
  // ---- conv3: 192->256, BN, ReLU ----  tiles = 512*4
  gemm_wmma_kernel<<<(512 * 4 + 7) / 8, 256, 0, stream>>>(concatB, 192, c3p, c3b, convbuf,
                                                          NPTS, 192, 256);
  bn_stats_kernel<<<256, 256, 0, stream>>>(convbuf, NPTS, 256, mu, inv);
  bn_relu_bf16_kernel<<<(NPTS * 256 + 255) / 256, 256, 0, stream>>>(convbuf, bn3g, bn3b, mu,
                                                                    inv, h3, NPTS, 256, 256, 0);
  // ---- conv4: 256->1024, BN, ReLU -> d_out transposed [1024, N] ----  tiles = 512*16
  gemm_wmma_kernel<<<(512 * 16 + 7) / 8, 256, 0, stream>>>(h3, 256, c4p, c4b, convbuf, NPTS,
                                                           256, 1024);
  bn_stats_kernel<<<1024, 256, 0, stream>>>(convbuf, NPTS, 1024, mu, inv);
  bn_relu_out_kernel<<<(NPTS * 1024 + 255) / 256, 256, 0, stream>>>(convbuf, bn4g, bn4b, mu,
                                                                    inv, out, NPTS, 1024);
}
